// GlobalSelfAttention_10264971837810
// MI455X (gfx1250) — matrix operand order repaired and checked by reading, not compiled, not run on hardware
//
#include <hip/hip_runtime.h>

// ---------------------------------------------------------------------------
// Problem constants (match reference)
// ---------------------------------------------------------------------------
#define N_NODES   4096
#define HID       256
#define NHEADS    8
#define HDIM      32
#define NPAD      4128            // Vt column padding so j-chunks never read OOB
#define ATT_SCALE 0.17677669529663687f   // HDIM^-0.5
#define LN_EPS    1e-5f

typedef _Float16 f16;
typedef __attribute__((ext_vector_type(8)))  f16   v8h;
typedef __attribute__((ext_vector_type(16))) f16   v16h;
typedef __attribute__((ext_vector_type(8)))  float v8f;

// ---------------------------------------------------------------------------
// Fragment helpers (CDNA5 WMMA 16x16x32 f16, wave32; k order used here is contiguous, k = 16*(L/16) + s)
//  A (16x32): lane L -> row = L%16; halfwords 0..7  = K[(L/16)*16 + 0..7]
//                                   halfwords 8..15 = K[(L/16)*16 + 8..15]
//  B (32x16): lane L -> col = L%16; halfwords 0..15 = K[(L/16)*16 + 0..15]
//  C (16x16 f32): lane L -> col = L%16; VGPR r -> row = r + 8*(L/16)
// ---------------------------------------------------------------------------
__device__ inline v16h join16(v8h lo, v8h hi) {
  v16h r;
#pragma unroll
  for (int i = 0; i < 8; ++i) { r[i] = lo[i]; r[i + 8] = hi[i]; }
  return r;
}
// 16 contiguous halfwords (B fragment: K-run of 16)
__device__ inline v16h load_c16(const f16* p) {
  const v8h* q = (const v8h*)p;
  return join16(q[0], q[1]);
}
// two 8-halfword runs (A fragment: K[half*16+0..7] and K[half*16+8..15])
__device__ inline v16h load_split(const f16* p0, const f16* p1) {
  return join16(*(const v8h*)p0, *(const v8h*)p1);
}
__device__ inline v8f zero8() {
  v8f z;
#pragma unroll
  for (int i = 0; i < 8; ++i) z[i] = 0.0f;
  return z;
}
__device__ inline v8f wmma_f16(v16h a, v16h b, v8f c) {
  // (neg_a, A, neg_b, B, c_mod, C, reuse_a, reuse_b)
  return __builtin_amdgcn_wmma_f32_16x16x32_f16(false, a, false, b, (short)0, c,
                                                false, false);
}

// ---------------------------------------------------------------------------
// Kernel 1: per-row same-graph segment bounds (batch is sorted int64)
// ---------------------------------------------------------------------------
__global__ void seg_bounds_kernel(const long long* __restrict__ batch,
                                  int* __restrict__ row_lo,
                                  int* __restrict__ row_hi) {
  int i = blockIdx.x * blockDim.x + threadIdx.x;
  if (i >= N_NODES) return;
  long long b = batch[i];
  int l = 0, r = N_NODES;
  while (l < r) { int m = (l + r) >> 1; if (batch[m] < b) l = m + 1; else r = m; }
  row_lo[i] = l;
  l = 0; r = N_NODES;
  while (l < r) { int m = (l + r) >> 1; if (batch[m] <= b) l = m + 1; else r = m; }
  row_hi[i] = l;
}

// ---------------------------------------------------------------------------
// Kernel 2: LayerNorm -> f16   (one wave32 per row, 8 rows per block)
// ---------------------------------------------------------------------------
__global__ void ln_f16_kernel(const float* __restrict__ x,
                              const float* __restrict__ gamma,
                              const float* __restrict__ beta,
                              f16* __restrict__ h) {
  const int lane = threadIdx.x & 31;
  const int wave = threadIdx.x >> 5;
  const int row  = blockIdx.x * 8 + wave;
  if (row >= N_NODES) return;
  const float* xr = x + (size_t)row * HID;
  float v[8];
  float s = 0.0f;
#pragma unroll
  for (int k = 0; k < 8; ++k) { v[k] = xr[lane + 32 * k]; s += v[k]; }
#pragma unroll
  for (int off = 16; off > 0; off >>= 1) s += __shfl_xor(s, off, 32);
  const float mu = s * (1.0f / HID);
  float q = 0.0f;
#pragma unroll
  for (int k = 0; k < 8; ++k) { float d = v[k] - mu; q += d * d; }
#pragma unroll
  for (int off = 16; off > 0; off >>= 1) q += __shfl_xor(q, off, 32);
  const float inv = rsqrtf(q * (1.0f / HID) + LN_EPS);
#pragma unroll
  for (int k = 0; k < 8; ++k) {
    int c = lane + 32 * k;
    h[(size_t)row * HID + c] = (f16)((v[k] - mu) * inv * gamma[c] + beta[c]);
  }
}

// ---------------------------------------------------------------------------
// Kernel 3: weight convert+transpose  W[in][out] f32 -> Wt[out][in] f16
// ---------------------------------------------------------------------------
__global__ void wtrans_kernel(const float* __restrict__ W, f16* __restrict__ Wt) {
  int idx = blockIdx.x * blockDim.x + threadIdx.x;   // idx = n*HID + k
  if (idx >= HID * HID) return;
  int k = idx & (HID - 1);
  int n = idx >> 8;
  Wt[idx] = (f16)W[(size_t)k * HID + n];
}

// ---------------------------------------------------------------------------
// Kernel 4: QKV projections.  One wave per 16x16 tile; z selects Q/K/V.
//   Q f16 [N][HID] (pre-scaled by ATT_SCALE), K f16 [N][HID],
//   V stored transposed: Vt f16 [HID][NPAD]  (channel-major for AV B-fragments)
// ---------------------------------------------------------------------------
__global__ void qkv_gemm_kernel(const f16* __restrict__ h,
                                const f16* __restrict__ WqT,
                                const f16* __restrict__ WkT,
                                const f16* __restrict__ WvT,
                                const float* __restrict__ bq,
                                const float* __restrict__ bk,
                                const float* __restrict__ bv,
                                f16* __restrict__ Qf,
                                f16* __restrict__ Kf,
                                f16* __restrict__ Vt) {
  const int lane  = threadIdx.x & 31;
  const int l16   = lane & 15;
  const int half  = lane >> 4;
  const int mbase = blockIdx.x * 16;
  const int nbase = blockIdx.y * 16;
  const int z     = blockIdx.z;

  const f16*   Wt   = (z == 0) ? WqT : (z == 1) ? WkT : WvT;
  const float* bias = (z == 0) ? bq  : (z == 1) ? bk  : bv;

  const f16* arow = h + (size_t)(mbase + l16) * HID;
  const f16* brow = Wt + (size_t)(nbase + l16) * HID;
  v8f acc = zero8();
#pragma unroll
  for (int kk = 0; kk < HID; kk += 32) {
    v16h a = load_split(arow + kk + half * 16, arow + kk + half * 16 + 8);
    v16h b = load_c16(brow + kk + half * 16);
    acc = wmma_f16(a, b, acc);
  }
  const int n = nbase + l16;
  const float bn = bias[n];
#pragma unroll
  for (int r = 0; r < 8; ++r) {
    int m = mbase + r + 8 * half;
    float val = acc[r] + bn;
    if (z == 0)      Qf[(size_t)m * HID + n] = (f16)(val * ATT_SCALE);
    else if (z == 1) Kf[(size_t)m * HID + n] = (f16)val;
    else             Vt[(size_t)n * NPAD + m] = (f16)val;
  }
}

// ---------------------------------------------------------------------------
// Kernel 5: block-diagonal flash attention.
//   grid = (N/16 query tiles, NHEADS), one wave per block.
//   Online softmax over the tile's union key range; per-row [lo,hi) mask.
// ---------------------------------------------------------------------------
__global__ void attn_kernel(const f16* __restrict__ Qf,
                            const f16* __restrict__ Kf,
                            const f16* __restrict__ Vt,
                            const int* __restrict__ row_lo,
                            const int* __restrict__ row_hi,
                            f16* __restrict__ O) {
  __shared__ f16 Plds[16][32];    // P tile staging: C-layout -> A-layout
  const int lane  = threadIdx.x & 31;
  const int l16   = lane & 15;
  const int half  = lane >> 4;
  const int head  = blockIdx.y;
  const int ibase = blockIdx.x * 16;

  // per-lane rows are (r + 8*half); gather their mask ranges + tile union
  int lo_r[8], hi_r[8];
  int tlo = N_NODES, thi = 0;
#pragma unroll
  for (int r = 0; r < 8; ++r) {
    int gi = ibase + r + 8 * half;
    lo_r[r] = row_lo[gi];
    hi_r[r] = row_hi[gi];
    tlo = lo_r[r] < tlo ? lo_r[r] : tlo;
    thi = hi_r[r] > thi ? hi_r[r] : thi;
  }
#pragma unroll
  for (int off = 16; off > 0; off >>= 1) {
    int t = __shfl_xor(tlo, off, 32); tlo = t < tlo ? t : tlo;
    t = __shfl_xor(thi, off, 32);     thi = t > thi ? t : thi;
  }

  // Q A-fragment for this tile/head (Q pre-scaled by ATT_SCALE)
  const f16* qrow = Qf + (size_t)(ibase + l16) * HID + head * HDIM;
  const v16h aq = load_split(qrow + half * 16, qrow + half * 16 + 8);

  v8f o0 = zero8(), o1 = zero8();
  float m_r[8], l_r[8];
#pragma unroll
  for (int r = 0; r < 8; ++r) { m_r[r] = -__builtin_inff(); l_r[r] = 0.0f; }

  for (int jb = (tlo & ~15); jb < thi; jb += 32) {
    // --- scores: S = Q (16x32) x K^T (32 x 32cols) as two 16-col WMMAs ---
    const int j0 = jb + l16;
    const int j1 = jb + 16 + l16;
    const int j0c = j0 < (N_NODES - 1) ? j0 : (N_NODES - 1);
    const int j1c = j1 < (N_NODES - 1) ? j1 : (N_NODES - 1);
    v16h kb0 = load_c16(Kf + (size_t)j0c * HID + head * HDIM + half * 16);
    v16h kb1 = load_c16(Kf + (size_t)j1c * HID + head * HDIM + half * 16);
    v8f s0 = wmma_f16(aq, kb0, zero8());
    v8f s1 = wmma_f16(aq, kb1, zero8());

    // --- masked online softmax update (per-row stats, 16-lane reductions) ---
    float p0[8], p1[8];
#pragma unroll
    for (int r = 0; r < 8; ++r) {
      const bool v0 = (j0 >= lo_r[r]) & (j0 < hi_r[r]);
      const bool v1 = (j1 >= lo_r[r]) & (j1 < hi_r[r]);
      float x0 = v0 ? s0[r] : -__builtin_inff();
      float x1 = v1 ? s1[r] : -__builtin_inff();
      float mx = fmaxf(x0, x1);
#pragma unroll
      for (int off = 8; off > 0; off >>= 1) mx = fmaxf(mx, __shfl_xor(mx, off, 16));
      const float mnew = fmaxf(m_r[r], mx);
      const float scale =
          (m_r[r] > -__builtin_inff()) ? __expf(m_r[r] - mnew) : 0.0f;
      p0[r] = v0 ? __expf(s0[r] - mnew) : 0.0f;
      p1[r] = v1 ? __expf(s1[r] - mnew) : 0.0f;
      float sum = p0[r] + p1[r];
#pragma unroll
      for (int off = 8; off > 0; off >>= 1) sum += __shfl_xor(sum, off, 16);
      l_r[r] = l_r[r] * ((m_r[r] > -__builtin_inff()) ? scale : 1.0f) + sum;
      m_r[r] = mnew;
      o0[r] *= scale;
      o1[r] *= scale;
    }

    // --- transpose P (C layout, lane=col) -> A fragment (lane=row) via LDS ---
    __syncthreads();
#pragma unroll
    for (int r = 0; r < 8; ++r) {
      Plds[r + 8 * half][l16]      = (f16)p0[r];
      Plds[r + 8 * half][16 + l16] = (f16)p1[r];
    }
    __syncthreads();
    v16h ap = load_split(&Plds[l16][half * 16], &Plds[l16][half * 16 + 8]);

    // --- accumulate O += P (16x32) x V (32 x 32cols of head) ---
    v16h vb0 = load_c16(Vt + (size_t)(head * HDIM + l16)      * NPAD + jb + half * 16);
    v16h vb1 = load_c16(Vt + (size_t)(head * HDIM + 16 + l16) * NPAD + jb + half * 16);
    o0 = wmma_f16(ap, vb0, o0);
    o1 = wmma_f16(ap, vb1, o1);
  }

  // epilogue: normalize by denominator, store O f16 [N][HID]
#pragma unroll
  for (int r = 0; r < 8; ++r) {
    const int m = ibase + r + 8 * half;
    const float inv = (l_r[r] > 0.0f) ? (1.0f / l_r[r]) : 0.0f;
    O[(size_t)m * HID + head * HDIM + l16]      = (f16)(o0[r] * inv);
    O[(size_t)m * HID + head * HDIM + 16 + l16] = (f16)(o1[r] * inv);
  }
}

// ---------------------------------------------------------------------------
// Kernel 6: output projection + bias + residual -> f32 d_out
// ---------------------------------------------------------------------------
__global__ void out_proj_kernel(const f16* __restrict__ O,
                                const f16* __restrict__ WoT,
                                const float* __restrict__ bo,
                                const float* __restrict__ x,
                                float* __restrict__ out) {
  const int lane  = threadIdx.x & 31;
  const int l16   = lane & 15;
  const int half  = lane >> 4;
  const int mbase = blockIdx.x * 16;
  const int nbase = blockIdx.y * 16;

  const f16* arow = O + (size_t)(mbase + l16) * HID;
  const f16* brow = WoT + (size_t)(nbase + l16) * HID;
  v8f acc = zero8();
#pragma unroll
  for (int kk = 0; kk < HID; kk += 32) {
    v16h a = load_split(arow + kk + half * 16, arow + kk + half * 16 + 8);
    v16h b = load_c16(brow + kk + half * 16);
    acc = wmma_f16(a, b, acc);
  }
  const int n = nbase + l16;
  const float bn = bo[n];
#pragma unroll
  for (int r = 0; r < 8; ++r) {
    int m = mbase + r + 8 * half;
    out[(size_t)m * HID + n] = acc[r] + bn + x[(size_t)m * HID + n];
  }
}

// ---------------------------------------------------------------------------
// Launch
// ---------------------------------------------------------------------------
static inline size_t align256(size_t v) { return (v + 255) & ~(size_t)255; }

extern "C" void kernel_launch(void* const* d_in, const int* in_sizes, int n_in,
                              void* d_out, int out_size, void* d_ws, size_t ws_size,
                              hipStream_t stream) {
  const float*     x     = (const float*)d_in[0];
  const long long* batch = (const long long*)d_in[1];
  const float*     Wq    = (const float*)d_in[2];
  const float*     bq    = (const float*)d_in[3];
  const float*     Wk    = (const float*)d_in[4];
  const float*     bk    = (const float*)d_in[5];
  const float*     Wv    = (const float*)d_in[6];
  const float*     bv    = (const float*)d_in[7];
  const float*     Wo    = (const float*)d_in[8];
  const float*     bo    = (const float*)d_in[9];
  const float*     gamma = (const float*)d_in[10];
  const float*     beta  = (const float*)d_in[11];
  float*           out   = (float*)d_out;

  // scratch carve
  char* ws = (char*)d_ws;
  size_t off = 0;
  int* row_lo = (int*)(ws + off); off = align256(off + N_NODES * sizeof(int));
  int* row_hi = (int*)(ws + off); off = align256(off + N_NODES * sizeof(int));
  f16* h   = (f16*)(ws + off); off = align256(off + (size_t)N_NODES * HID * 2);
  f16* Qf  = (f16*)(ws + off); off = align256(off + (size_t)N_NODES * HID * 2);
  f16* Kf  = (f16*)(ws + off); off = align256(off + (size_t)N_NODES * HID * 2);
  f16* Of  = (f16*)(ws + off); off = align256(off + (size_t)N_NODES * HID * 2);
  f16* Vt  = (f16*)(ws + off); off = align256(off + (size_t)HID * NPAD * 2);
  f16* WqT = (f16*)(ws + off); off = align256(off + (size_t)HID * HID * 2);
  f16* WkT = (f16*)(ws + off); off = align256(off + (size_t)HID * HID * 2);
  f16* WvT = (f16*)(ws + off); off = align256(off + (size_t)HID * HID * 2);
  f16* WoT = (f16*)(ws + off); off = align256(off + (size_t)HID * HID * 2);

  seg_bounds_kernel<<<N_NODES / 256, 256, 0, stream>>>(batch, row_lo, row_hi);
  ln_f16_kernel<<<N_NODES / 8, 256, 0, stream>>>(x, gamma, beta, h);
  wtrans_kernel<<<(HID * HID) / 256, 256, 0, stream>>>(Wq, WqT);
  wtrans_kernel<<<(HID * HID) / 256, 256, 0, stream>>>(Wk, WkT);
  wtrans_kernel<<<(HID * HID) / 256, 256, 0, stream>>>(Wv, WvT);
  wtrans_kernel<<<(HID * HID) / 256, 256, 0, stream>>>(Wo, WoT);
  qkv_gemm_kernel<<<dim3(N_NODES / 16, HID / 16, 3), 32, 0, stream>>>(
      h, WqT, WkT, WvT, bq, bk, bv, Qf, Kf, Vt);
  attn_kernel<<<dim3(N_NODES / 16, NHEADS), 32, 0, stream>>>(
      Qf, Kf, Vt, row_lo, row_hi, Of);
  out_proj_kernel<<<dim3(N_NODES / 16, HID / 16), 32, 0, stream>>>(
      Of, WoT, bo, x, out);
}
